// GraphAttentionLayer_32555852103747
// MI455X (gfx1250) — matrix operand order repaired and checked by reading, not compiled, not run on hardware
//
#include <hip/hip_runtime.h>
#include <hip/hip_bf16.h>

// GAT layer fused kernel set for gfx1250 (MI455X), wave32.
// N=8192, F_IN=128, F_OUT=64. Output f32 (8192x64).
//
// Workspace layout (P = j-split parts, chosen from ws_size):
//   [0, P*2MiB)        partial acc f32 [P][8192][64]   (part 0 overlays Wh)
//   [0, 2MiB)          Wh f32 row-major (live only gemm -> rowops)
//   [P*2MiB, +1MiB)    WhhT f16 column-major [64][8192]
//   then f1 (32KB), f2 (32KB), f2max key (4KB pad), lparts f32 [P][8192]

#define NROWS  8192
#define FIN    128
#define FOUT   64
#define NEG_SLOPE 0.01f
#define L2E    1.44269504088896340736f

typedef __attribute__((ext_vector_type(2)))  float    v2f;
typedef __attribute__((ext_vector_type(8)))  float    v8f;
typedef __attribute__((ext_vector_type(16))) _Float16 v16h;
typedef __attribute__((ext_vector_type(4)))  int      v4i;

#if defined(__gfx1250__) && __has_builtin(__builtin_amdgcn_global_load_async_to_lds_b128)
#define USE_ASYNC_LDS 1
typedef __attribute__((address_space(1))) v4i* gptr_v4i;   // global int4*
typedef __attribute__((address_space(3))) v4i* lptr_v4i;   // LDS int4*
#else
#define USE_ASYNC_LDS 0
#endif

__device__ __forceinline__ void wait_async0() {
#if defined(__gfx1250__)
#if __has_builtin(__builtin_amdgcn_s_wait_asynccnt)
    __builtin_amdgcn_s_wait_asynccnt(0);
#else
    asm volatile("s_wait_asynccnt 0x0" ::: "memory");
#endif
#endif
}

__device__ __forceinline__ float fast_exp2(float x) {
#if defined(__gfx1250__) && __has_builtin(__builtin_amdgcn_exp2f)
    return __builtin_amdgcn_exp2f(x);     // v_exp_f32
#else
    return exp2f(x);
#endif
}

__device__ __forceinline__ unsigned enc_f32(float x) {
    unsigned u = __float_as_uint(x);
    return (u & 0x80000000u) ? ~u : (u | 0x80000000u);   // order-preserving
}
__device__ __forceinline__ float dec_f32(unsigned k) {
    unsigned u = (k & 0x80000000u) ? (k & 0x7fffffffu) : ~k;
    return __uint_as_float(u);
}

__global__ void gat_init_kernel(unsigned* f2key) {
    if (threadIdx.x == 0) *f2key = 0u;   // < encoding of every real float
}

// ---------------------------------------------------------------------------
// Kernel 1: Wh = h @ W with V_WMMA_F32_16X16X4_F32 (full fp32 precision).
// Stores Wh f32 row-major and WhhT f16 column-major (packed b128 store).
// ---------------------------------------------------------------------------
__global__ void gat_gemm_wh(const float* __restrict__ h,
                            const float* __restrict__ W,
                            float* __restrict__ Wh,
                            _Float16* __restrict__ WhhT) {
    const int wv    = (blockIdx.x * blockDim.x + threadIdx.x) >> 5;
    const int lane  = threadIdx.x & 31;
    const int nlane = lane & 15;
    const int hi    = lane >> 4;
    const int tM = (wv >> 2) * 16;        // 512 M-tiles
    const int tN = (wv & 3) * 16;         // 4 N-tiles

    v8f acc = {};
    const int rowA = tM + nlane;
#pragma unroll 4
    for (int kb = 0; kb < FIN / 4; ++kb) {
        const int kA = kb * 4 + hi * 2;
        float2 av = *(const float2*)(h + (size_t)rowA * FIN + kA);
        v2f a2; a2[0] = av.x; a2[1] = av.y;
        v2f b2;
        b2[0] = W[(size_t)kA * FOUT + tN + nlane];
        b2[1] = W[(size_t)(kA + 1) * FOUT + tN + nlane];
        acc = __builtin_amdgcn_wmma_f32_16x16x4_f32(
                  false, a2, false, b2, (short)0, acc, false, false);
    }

    const int col = tN + nlane;
    union { _Float16 hv[8]; uint4 u; } cv;
#pragma unroll
    for (int r = 0; r < 8; ++r) {
        const int row = tM + r + hi * 8;      // C/D layout: VGPR r -> M=r / M=r+8
        const float v = acc[r];
        Wh[(size_t)row * FOUT + col] = v;
        cv.hv[r] = (_Float16)v;
    }
    *(uint4*)(WhhT + (size_t)col * NROWS + tM + hi * 8) = cv.u;
}

// ---------------------------------------------------------------------------
// Kernel 2: f1[i] = Wh[i]·a1, f2[i] = Wh[i]·a2, global max(f2) via atomic.
// ---------------------------------------------------------------------------
__global__ void gat_rowops(const float* __restrict__ Wh,
                           const float* __restrict__ a,
                           float* __restrict__ f1,
                           float* __restrict__ f2,
                           unsigned* __restrict__ f2key) {
    const int row  = blockIdx.x * 8 + (threadIdx.x >> 5);
    const int lane = threadIdx.x & 31;
    const float w0 = Wh[(size_t)row * FOUT + lane];
    const float w1 = Wh[(size_t)row * FOUT + 32 + lane];
    float v1 = w0 * a[lane]      + w1 * a[32 + lane];
    float v2 = w0 * a[64 + lane] + w1 * a[96 + lane];
#pragma unroll
    for (int off = 16; off > 0; off >>= 1) {
        v1 += __shfl_down(v1, off);
        v2 += __shfl_down(v2, off);
    }
    if (lane == 0) {
        f1[row] = v1;
        f2[row] = v2;
        atomicMax(f2key, enc_f32(v2));
    }
}

// ---------------------------------------------------------------------------
// Kernel 3: fused attention over a j-slice.  grid = 128*parts blocks x 128.
// Wave owns 16 rows x 64 cols for one j-slice.  Per 32-wide tile:
//   p = exp2( max(x + A1, fma(.01, x, A2)) ),  x = f2*log2e (pre-scaled LDS)
//   acc[0..3] += P @ WhTile  (4x V_WMMA_F32_16X16X32_F16, f32 accum)
//   acc[4]    += P @ ones    (row sums land in every lane's own rows)
// Tile double-buffered in LDS via GLOBAL_LOAD_ASYNC_TO_LDS_B128 + asynccnt.
// Writes unnormalized partials; gat_combine normalizes + elu.
// ---------------------------------------------------------------------------
__device__ __forceinline__ void stage_tile(const _Float16* __restrict__ WhhT,
                                           _Float16* buf, int j0, int tid) {
#pragma unroll
    for (int s = tid; s < 256; s += 128) {      // 256 slots of 16B
        const int n = s >> 2, q = s & 3;
        const _Float16* gp = WhhT + (size_t)n * NROWS + j0 + q * 8;
        _Float16* lp = buf + n * 32 + q * 8;
#if USE_ASYNC_LDS
        __builtin_amdgcn_global_load_async_to_lds_b128(
            (gptr_v4i)(void*)gp, (lptr_v4i)(void*)lp, 0, 0);
#else
        *(uint4*)lp = *(const uint4*)gp;
#endif
    }
}

__global__ void __launch_bounds__(128)
gat_attention(const _Float16* __restrict__ WhhT,
              const float* __restrict__ f1,
              const float* __restrict__ f2,
              const unsigned* __restrict__ f2key,
              float* __restrict__ accs,      // [parts][8192][64]
              float* __restrict__ lparts,    // [parts][8192]
              int tilesPerPart) {
    __shared__ float f2s[NROWS];                               // <= 32 KB used
    __shared__ __align__(64) _Float16 whT[2][FOUT * 32];       // 8 KB

    const int tid   = threadIdx.x;
    const int wv    = tid >> 5;
    const int lane  = tid & 31;
    const int nlane = lane & 15;
    const int hi    = lane >> 4;
    const int hi8   = hi * 8;
    const int part   = blockIdx.x >> 7;
    const int rowblk = blockIdx.x & 127;
    const int i0    = rowblk * 64 + wv * 16;       // this wave's 16 rows
    const int jbase = part * tilesPerPart * 32;    // this part's j-slice
    const int jlen  = tilesPerPart * 32;

    // stage this slice of f2, pre-scaled by log2(e)
    for (int idx = tid; idx < jlen; idx += 128)
        f2s[idx] = f2[jbase + idx] * L2E;

    const float f2max = dec_f32(*f2key);
    const float f1v   = f1[i0 + nlane];
    float m = f1v + f2max;
    m = m > 0.f ? m : NEG_SLOPE * m;               // exact row max of e
    const float A1 = (f1v - m) * L2E;              // pos-branch offset
    const float A2 = (NEG_SLOPE * f1v - m) * L2E;  // neg-branch offset
    const float K2 = NEG_SLOPE;                    // slope on pre-scaled x

    v16h bones;                                    // ones B-matrix (row sums)
#pragma unroll
    for (int e = 0; e < 16; ++e) bones[e] = (_Float16)1.0f;

    v8f acc[5] = {};                               // 4 N-tiles + row-sum tile

    stage_tile(WhhT, whT[0], jbase, tid);          // prologue: tile 0 in flight

    for (int jt = 0; jt < tilesPerPart; ++jt) {
#if USE_ASYNC_LDS
        wait_async0();          // my async writes for tile jt have landed
#endif
        __syncthreads();        // everyone's tile jt visible; other buf free

        if (jt + 1 < tilesPerPart)
            stage_tile(WhhT, whT[(jt + 1) & 1], jbase + jt * 32 + 32, tid);

        const _Float16* cur = whT[jt & 1];

        // A fragment: element e -> K = e + hi*16
        v16h afrag;
#pragma unroll
        for (int e = 0; e < 16; ++e) {
            const int k = jt * 32 + e + hi * 16;
            const float x  = f2s[k];
            const float t1 = x + A1;               // log2e*(f1+f2-m)
            const float t2 = __builtin_fmaf(K2, x, A2);
            afrag[e] = (_Float16)fast_exp2(fmaxf(t1, t2));
        }

        // B fragment lane c: K = e + hi*16, N = 16t + nlane
#pragma unroll
        for (int t = 0; t < 4; ++t) {
            const v16h b = *(const v16h*)&cur[(t * 16 + nlane) * 32 + hi * 16];
            acc[t] = __builtin_amdgcn_wmma_f32_16x16x32_f16(
                         false, afrag, false, b, (short)0, acc[t], false, false);
        }
        acc[4] = __builtin_amdgcn_wmma_f32_16x16x32_f16(
                     false, afrag, false, bones, (short)0, acc[4], false, false);
    }

    // store unnormalized partials; C/D layout: VGPR r -> row r + hi*8
    float* accp = accs + (size_t)part * NROWS * FOUT;
#pragma unroll
    for (int r = 0; r < 8; ++r) {
        const int row = i0 + r + hi8;
#pragma unroll
        for (int t = 0; t < 4; ++t)
            accp[(size_t)row * FOUT + t * 16 + nlane] = acc[t][r];
    }
    if (nlane == 0) {          // lanes 0 & 16 carry the row sums for rows +hi8
#pragma unroll
        for (int r = 0; r < 8; ++r)
            lparts[(size_t)part * NROWS + i0 + r + hi8] = acc[4][r];
    }
}

// ---------------------------------------------------------------------------
// Kernel 4: out = elu( sum_p acc_p / sum_p l_p )
// ---------------------------------------------------------------------------
__global__ void gat_combine(const float* __restrict__ accs,
                            const float* __restrict__ lparts,
                            float* __restrict__ out, int parts) {
    const int idx = blockIdx.x * 256 + threadIdx.x;   // 0 .. 8192*64-1
    const int row = idx >> 6;
    float s = 0.f, l = 0.f;
    for (int p = 0; p < parts; ++p) {
        s += accs[(size_t)p * NROWS * FOUT + idx];
        l += lparts[(size_t)p * NROWS + row];
    }
    float v = s / l;
    out[idx] = v > 0.f ? v : (__expf(v) - 1.f);       // elu (alpha=1)
}

extern "C" void kernel_launch(void* const* d_in, const int* in_sizes, int n_in,
                              void* d_out, int out_size, void* d_ws, size_t ws_size,
                              hipStream_t stream) {
    const float* h = (const float*)d_in[0];   // 8192*128
    const float* W = (const float*)d_in[1];   // 128*64
    const float* a = (const float*)d_in[2];   // 128
    float* out = (float*)d_out;               // 8192*64

    // choose j-split (deterministic: depends only on ws_size)
    const size_t ACC1 = (size_t)NROWS * FOUT * 4;             // 2 MiB per part
    auto need = [&](int p) {
        return (size_t)p * ACC1 + (size_t)NROWS * FOUT * 2    // WhhT
             + 2 * (size_t)NROWS * 4 + 4096                   // f1,f2,key
             + (size_t)p * NROWS * 4;                         // lparts
    };
    int parts = (ws_size >= need(4)) ? 4 : (ws_size >= need(2)) ? 2 : 1;

    char* ws = (char*)d_ws;
    float*    accs  = (float*)ws;                        // [parts][8192][64]
    float*    Wh    = (float*)ws;                        // overlays part 0 (dead after rowops)
    _Float16* WhhT  = (_Float16*)(ws + (size_t)parts * ACC1);
    char*     tail  = ws + (size_t)parts * ACC1 + (size_t)NROWS * FOUT * 2;
    float*    f1    = (float*)tail;
    float*    f2    = (float*)(tail + NROWS * 4);
    unsigned* f2key = (unsigned*)(tail + 2 * NROWS * 4);
    float*    lpart = (float*)(tail + 2 * NROWS * 4 + 4096);

    gat_init_kernel<<<1, 32, 0, stream>>>(f2key);
    gat_gemm_wh<<<512, 128, 0, stream>>>(h, W, Wh, WhhT);
    gat_rowops<<<NROWS / 8, 256, 0, stream>>>(Wh, a, f1, f2, f2key);
    gat_attention<<<128 * parts, 128, 0, stream>>>(WhhT, f1, f2, f2key,
                                                   accs, lpart,
                                                   (NROWS / 32) / parts);
    gat_combine<<<NROWS * FOUT / 256, 256, 0, stream>>>(accs, lpart, out, parts);
}
